// TreeBertLayer_71579924955680
// MI455X (gfx1250) — matrix operand order repaired and checked by reading, not compiled, not run on hardware
//
#include <hip/hip_runtime.h>
#include <hip/hip_bf16.h>
#include <math.h>

#define BB  16
#define SS  512
#define DD  768
#define HH  12
#define DKK 64
#define DII 3072
#define NEGV (-1e9f)

typedef __attribute__((ext_vector_type(16))) __bf16 bf16x16;
typedef __attribute__((ext_vector_type(8)))  float  f32x8;
typedef __attribute__((ext_vector_type(4)))  int    i32x4;

// --------------------------- CDNA5 async-LDS path ---------------------------
#if defined(__gfx1250__) && __has_builtin(__builtin_amdgcn_global_load_async_to_lds_b128)
#define HAVE_ASYNC_LDS 1
#else
#define HAVE_ASYNC_LDS 0
#endif

// 16-byte global -> LDS copy (async DMA when available, register copy else).
__device__ __forceinline__ void cp16(__bf16* l, const __bf16* g) {
#if HAVE_ASYNC_LDS
  i32x4* gg = (i32x4*)(void*)const_cast<__bf16*>(g);
  i32x4* ll = (i32x4*)(void*)l;
  __builtin_amdgcn_global_load_async_to_lds_b128(
      (__attribute__((address_space(1))) i32x4*)gg,
      (__attribute__((address_space(3))) i32x4*)ll, 0, 0);
#else
  *(float4*)(void*)l = *(const float4*)(const void*)g;
#endif
}

__device__ __forceinline__ void wait_async_le4() {
#if HAVE_ASYNC_LDS
#if __has_builtin(__builtin_amdgcn_s_wait_asynccnt)
  __builtin_amdgcn_s_wait_asynccnt(4);
#else
  asm volatile("s_wait_asynccnt 0x4" ::: "memory");
#endif
#endif
}
__device__ __forceinline__ void wait_async_le0() {
#if HAVE_ASYNC_LDS
#if __has_builtin(__builtin_amdgcn_s_wait_asynccnt)
  __builtin_amdgcn_s_wait_asynccnt(0);
#else
  asm volatile("s_wait_asynccnt 0x0" ::: "memory");
#endif
#endif
}

// ------------------------------- WMMA helpers -------------------------------
__device__ __forceinline__ f32x8 wmma_bf16(bf16x16 a, bf16x16 b, f32x8 c) {
  return __builtin_amdgcn_wmma_f32_16x16x32_bf16(false, a, false, b, (short)0, c,
                                                 false, false);
}

// A fragment: 16x32 (MxK). lane L: row = base + L%16, half = L/16.
// elements 0..7 -> K = half*8+j ; elements 8..15 -> K = 16 + half*8 + j
__device__ __forceinline__ bf16x16 frag_a(const __bf16* tile, int ld, int row,
                                          int half, int k0) {
  const __bf16* p = tile + (size_t)row * ld + k0;
  bf16x16 a;
#pragma unroll
  for (int j = 0; j < 8; ++j) {
    a[j]     = p[half * 8 + j];
    a[8 + j] = p[16 + half * 8 + j];
  }
  return a;
}
// B fragment from LDS tile stored [N][K]: lane L: col = base + L%16,
// elements 0..7 -> K = half*8+j ; elements 8..15 -> K = 16 + half*8 + j
__device__ __forceinline__ bf16x16 frag_b(const __bf16* tileT, int ld, int col,
                                          int half, int k0) {
  const __bf16* p = tileT + (size_t)col * ld + k0;
  bf16x16 b;
#pragma unroll
  for (int j = 0; j < 8; ++j) {
    b[j]     = p[half * 8 + j];
    b[8 + j] = p[16 + half * 8 + j];
  }
  return b;
}

// ---------------------------------------------------------------------------
// One-time prep: fp32 -> bf16 elementwise (x), and W[K][N] fp32 -> Wt[N][K] bf16
// ---------------------------------------------------------------------------
__global__ __launch_bounds__(256) void f32_to_bf16_vec4(
    const float* __restrict__ X, __bf16* __restrict__ Y) {
  size_t i = ((size_t)blockIdx.x * 256 + threadIdx.x) * 4;
  float4 v = *(const float4*)(X + i);
  Y[i + 0] = (__bf16)v.x; Y[i + 1] = (__bf16)v.y;
  Y[i + 2] = (__bf16)v.z; Y[i + 3] = (__bf16)v.w;
}

__global__ __launch_bounds__(256) void transpose_to_bf16(
    const float* __restrict__ W, __bf16* __restrict__ Wt, int K, int N) {
  __shared__ float tile[32][33];
  const int n0 = blockIdx.x * 32, k0 = blockIdx.y * 32;
  const int tx = threadIdx.x & 31, ty = threadIdx.x >> 5;  // ty 0..7
#pragma unroll
  for (int i = ty; i < 32; i += 8)
    tile[i][tx] = W[(size_t)(k0 + i) * N + n0 + tx];
  __syncthreads();
#pragma unroll
  for (int i = ty; i < 32; i += 8)
    Wt[(size_t)(n0 + i) * K + k0 + tx] = (__bf16)tile[tx][i];
}

// ---------------------------------------------------------------------------
// GEMM: Out[M,N] = A[M,K](bf16) @ Wt[N,K]^T(bf16) + bias (+GELU) (+resid)
// Double-buffered LDS, async global->LDS copies, 8 waves, 128x128x32 tiles.
// Out (fp32) and Outb (bf16 mirror for downstream WMMA consumers) nullable.
// ---------------------------------------------------------------------------
__global__ __launch_bounds__(256) void gemm_bf16_wmma(
    const __bf16* __restrict__ A, const __bf16* __restrict__ Wt,
    const float* __restrict__ bias, const float* __restrict__ resid,
    float* __restrict__ Out, __bf16* __restrict__ Outb,
    int M, int N, int K, int gelu) {
  __shared__ __bf16 As[2][128][40];  // [m][k], 80B rows (16B aligned)
  __shared__ __bf16 Bs[2][128][40];  // [n][k]

  const int t = threadIdx.x;
  const int lane = t & 31, wave = t >> 5;
  const int wm = wave >> 2;  // 0..1 -> row offset wm*64
  const int wn = wave & 3;   // 0..3 -> col offset wn*32
  const int half = lane >> 4, ln = lane & 15;
  const int m0 = blockIdx.y * 128;
  const int n0 = blockIdx.x * 128;
  const int crow = t >> 1, cseg = (t & 1) * 16;  // 2x16B chunks per matrix

  f32x8 acc[4][2];
#pragma unroll
  for (int i = 0; i < 4; ++i)
#pragma unroll
    for (int j = 0; j < 2; ++j) {
      f32x8 z = {0.f, 0.f, 0.f, 0.f, 0.f, 0.f, 0.f, 0.f};
      acc[i][j] = z;
    }

  auto issue = [&](int p, int k0) {
    const __bf16* ga = A + (size_t)(m0 + crow) * K + k0 + cseg;
    cp16(&As[p][crow][cseg], ga);
    cp16(&As[p][crow][cseg + 8], ga + 8);
    const __bf16* gb = Wt + (size_t)(n0 + crow) * K + k0 + cseg;
    cp16(&Bs[p][crow][cseg], gb);
    cp16(&Bs[p][crow][cseg + 8], gb + 8);
  };

  const int KT = K / 32;
  issue(0, 0);
  for (int kt = 0; kt < KT; ++kt) {
    const int p = kt & 1;
    if (kt + 1 < KT) {
      issue(p ^ 1, (kt + 1) * 32);
      wait_async_le4();  // in-order completion => tile kt has landed
    } else {
      wait_async_le0();
    }
    __syncthreads();

    bf16x16 bf0 = frag_b(&Bs[p][0][0], 40, wn * 32 + ln, half, 0);
    bf16x16 bf1 = frag_b(&Bs[p][0][0], 40, wn * 32 + 16 + ln, half, 0);
#pragma unroll
    for (int rm = 0; rm < 4; ++rm) {
      bf16x16 af = frag_a(&As[p][0][0], 40, wm * 64 + rm * 16 + ln, half, 0);
      acc[rm][0] = wmma_bf16(af, bf0, acc[rm][0]);
      acc[rm][1] = wmma_bf16(af, bf1, acc[rm][1]);
    }
    __syncthreads();  // all waves done reading buffer p before it is refilled
  }

#pragma unroll
  for (int rm = 0; rm < 4; ++rm)
#pragma unroll
    for (int cn = 0; cn < 2; ++cn) {
      int col = n0 + wn * 32 + cn * 16 + ln;
      float bv = bias ? bias[col] : 0.0f;
#pragma unroll
      for (int r = 0; r < 8; ++r) {
        int row = m0 + wm * 64 + rm * 16 + half * 8 + r;
        float v = acc[rm][cn][r] + bv;
        if (gelu) v = 0.5f * v * (1.0f + erff(v * 0.70710678118654752f));
        if (resid) v += resid[(size_t)row * N + col];
        if (Out) Out[(size_t)row * N + col] = v;
        if (Outb) Outb[(size_t)row * N + col] = (__bf16)v;
      }
    }
}

// ---------------------------------------------------------------------------
// Row LayerNorm over D=768 (one wave per row). fp32 and/or bf16 outputs.
// ---------------------------------------------------------------------------
__global__ __launch_bounds__(256) void layernorm_rows(
    const float* __restrict__ X, const float* __restrict__ g,
    const float* __restrict__ bta, float* __restrict__ Yf,
    __bf16* __restrict__ Yb) {
  const int lane = threadIdx.x & 31, wave = threadIdx.x >> 5;
  const size_t row = (size_t)blockIdx.x * 8 + wave;
  const float* x = X + row * DD;
  float vbuf[24], s = 0.f, s2 = 0.f;
#pragma unroll
  for (int i = 0; i < 24; ++i) {
    float v = x[lane + i * 32];
    vbuf[i] = v; s += v; s2 += v * v;
  }
#pragma unroll
  for (int off = 16; off; off >>= 1) {
    s += __shfl_xor(s, off, 32);
    s2 += __shfl_xor(s2, off, 32);
  }
  float mu = s * (1.0f / DD);
  float var = s2 * (1.0f / DD) - mu * mu;
  float rstd = rsqrtf(var + 1e-12f);
#pragma unroll
  for (int i = 0; i < 24; ++i) {
    int d = lane + i * 32;
    float o = (vbuf[i] - mu) * rstd * g[d] + bta[d];
    if (Yf) Yf[row * DD + d] = o;
    if (Yb) Yb[row * DD + d] = (__bf16)o;
  }
}

// ---------------------------------------------------------------------------
// Group attention neighbor dots (bf16 inputs): q[i].k[i+1], q[i].k[i-1]
// ---------------------------------------------------------------------------
__global__ __launch_bounds__(256) void neighbor_dots(
    const __bf16* __restrict__ gq, const __bf16* __restrict__ gk,
    float* __restrict__ dplus, float* __restrict__ dminus) {
  const int lane = threadIdx.x & 31, wave = threadIdx.x >> 5;
  const int r = blockIdx.x * 8 + wave;
  const int i = r & (SS - 1);
  const bool hp = (i < SS - 1), hm = (i > 0);
  const __bf16* q = gq + (size_t)r * DD;
  const __bf16* kp = gk + (size_t)r * DD + DD;
  const __bf16* km = gk + (size_t)r * DD - DD;
  float sp = 0.f, sm = 0.f;
  for (int d = lane; d < DD; d += 32) {
    float qv = (float)q[d];
    if (hp) sp += qv * (float)kp[d];
    if (hm) sm += qv * (float)km[d];
  }
#pragma unroll
  for (int off = 16; off; off >>= 1) {
    sp += __shfl_xor(sp, off, 32);
    sm += __shfl_xor(sm, off, 32);
  }
  if (lane == 0) { dplus[r] = sp; dminus[r] = sm; }
}

// ---------------------------------------------------------------------------
// Per-batch neighbor softmax -> e (superdiag of neigh), prior blend,
// exclusive prefix sums of L = log(blend(e)+1e-9). One 512-thread block/batch.
// ---------------------------------------------------------------------------
__global__ __launch_bounds__(512) void group_scan(
    const float* __restrict__ dplus, const float* __restrict__ dminus,
    const int* __restrict__ amask, const float* __restrict__ prior,
    float* __restrict__ eout, float* __restrict__ Pout) {
  __shared__ float smp[SS], smm[SS], sc[SS];
  const int b = blockIdx.x, i = threadIdx.x;
  const size_t r = (size_t)b * SS + i;
  float sp = (i < SS - 1)
                 ? ((amask[b * SS + i + 1] != 0) ? dplus[r] * (1.0f / DD) : NEGV)
                 : -INFINITY;
  float sm = (i > 0)
                 ? ((amask[b * SS + i - 1] != 0) ? dminus[r] * (1.0f / DD) : NEGV)
                 : -INFINITY;
  float m = fmaxf(fmaxf(sp, sm), NEGV);
  int nb = (i > 0) + (i < SS - 1);
  float Z = __expf(sp - m) + __expf(sm - m) +
            (float)(SS - nb) * __expf(NEGV - m);
  smp[i] = __expf(sp - m) / Z;
  smm[i] = __expf(sm - m) / Z;
  __syncthreads();
  float eraw = 0.f, L = 0.f;
  if (i < SS - 1) {
    eraw = sqrtf(smp[i] * smm[i + 1] + 1e-9f);
    float pr = prior[((size_t)b * SS + i) * SS + i + 1];
    float eb = pr + (1.0f - pr) * eraw;
    L = __logf(eb + 1e-9f);
  }
  eout[r] = eraw;
  sc[i] = L;
  __syncthreads();
#pragma unroll
  for (int off = 1; off < SS; off <<= 1) {
    float v = (i >= off) ? sc[i - off] : 0.0f;
    __syncthreads();
    sc[i] += v;
    __syncthreads();
  }
  Pout[r] = (i == 0) ? 0.0f : sc[i - 1];
}

// ---------------------------------------------------------------------------
// Dense fill of C and neigh from e / P / prior.
// ---------------------------------------------------------------------------
__global__ __launch_bounds__(256) void fill_c_neigh(
    const float* __restrict__ prior, const float* __restrict__ e,
    const float* __restrict__ P, float* __restrict__ Cout,
    float* __restrict__ Nout) {
  const size_t idx = (size_t)blockIdx.x * 256 + threadIdx.x;
  const int j = (int)(idx & (SS - 1));
  const int i = (int)((idx >> 9) & (SS - 1));
  const int b = (int)(idx >> 18);
  float nraw;
  if (j == i + 1)      nraw = e[b * SS + i];
  else if (i == j + 1) nraw = e[b * SS + j];
  else                 nraw = 3.16227766017e-5f;  // sqrt(1e-9)
  float pr = prior[idx];
  float nv = pr + (1.0f - pr) * nraw;
  Nout[idx] = nv;
  float cv;
  if (i == j) {
    cv = nv;
  } else {
    int lo = i < j ? i : j, hi = i < j ? j : i;
    cv = __expf(P[b * SS + hi] - P[b * SS + lo]) + 1e-9f;
  }
  Cout[idx] = cv;
}

// ---------------------------------------------------------------------------
// Flash-style MHA, bf16 q/k/v, group-prob C weights the numerator only.
// ---------------------------------------------------------------------------
__global__ __launch_bounds__(256) void flash_mha(
    const __bf16* __restrict__ Q, const __bf16* __restrict__ Kx,
    const __bf16* __restrict__ Vx, const float* __restrict__ C,
    const int* __restrict__ amask, __bf16* __restrict__ Octx) {
  __shared__ __bf16 Qs[64][72];
  __shared__ __bf16 Ks[64][72];   // [key][d]
  __shared__ __bf16 Vt[64][72];   // [d][key] (transposed)
  __shared__ float  Sc[64][68];
  __shared__ __bf16 Ps[64][72];
  __shared__ float  rowscale[64];
  __shared__ float  rowl[64];

  const int t = threadIdx.x;
  const int lane = t & 31, wave = t >> 5;
  const int half = lane >> 4, ln = lane & 15;
  const int q0 = blockIdx.x * 64;
  const int hh = blockIdx.y;
  const int b = blockIdx.z;
  const int rt = wave >> 1;
  const int ct0 = (wave & 1) * 2;
  const int lr = t >> 2, lseg = (t & 3) * 16;

  {
    const __bf16* qp = Q + (size_t)(b * SS + q0 + lr) * DD + hh * DKK + lseg;
    cp16(&Qs[lr][lseg], qp);
    cp16(&Qs[lr][lseg + 8], qp + 8);
  }
  float mrow = -INFINITY, lrow = 0.0f;
  f32x8 oacc[2];
  {
    f32x8 z = {0.f, 0.f, 0.f, 0.f, 0.f, 0.f, 0.f, 0.f};
    oacc[0] = z; oacc[1] = z;
  }

  for (int kb = 0; kb < SS / 64; ++kb) {
    __syncthreads();  // previous iteration's readers of Ks/Vt/Ps are done
    const __bf16* kp = Kx + (size_t)(b * SS + kb * 64 + lr) * DD + hh * DKK + lseg;
    cp16(&Ks[lr][lseg], kp);
    cp16(&Ks[lr][lseg + 8], kp + 8);
    const __bf16* vp = Vx + (size_t)(b * SS + kb * 64 + lr) * DD + hh * DKK + lseg;
#pragma unroll
    for (int j = 0; j < 16; ++j) Vt[lseg + j][lr] = vp[j];
    wait_async_le0();
    __syncthreads();

    // S = Q K^T / sqrt(64)
#pragma unroll
    for (int c = 0; c < 2; ++c) {
      int ct = ct0 + c;
      f32x8 s = {0.f, 0.f, 0.f, 0.f, 0.f, 0.f, 0.f, 0.f};
#pragma unroll
      for (int kk = 0; kk < DKK; kk += 32) {
        bf16x16 af = frag_a(&Qs[0][0], 72, rt * 16 + ln, half, kk);
        bf16x16 bf = frag_b(&Ks[0][0], 72, ct * 16 + ln, half, kk);
        s = wmma_bf16(af, bf, s);
      }
#pragma unroll
      for (int r = 0; r < 8; ++r)
        Sc[rt * 16 + half * 8 + r][ct * 16 + ln] = s[r] * 0.125f;
    }
    __syncthreads();

    // online softmax (4 lanes per row), P = prob * C
    {
      float sv[16];
      float mloc = -INFINITY;
#pragma unroll
      for (int j = 0; j < 16; ++j) {
        int key = kb * 64 + lseg + j;
        bool ok = (amask[b * SS + key] != 0) || (key == q0 + lr);
        float s = ok ? Sc[lr][lseg + j] : NEGV;
        sv[j] = s;
        mloc = fmaxf(mloc, s);
      }
      mloc = fmaxf(mloc, __shfl_xor(mloc, 1, 4));
      mloc = fmaxf(mloc, __shfl_xor(mloc, 2, 4));
      float mnew = fmaxf(mrow, mloc);
      float ssum = 0.0f;
      const float* crow = C + (size_t)(b * SS + q0 + lr) * SS + kb * 64 + lseg;
#pragma unroll
      for (int j = 0; j < 16; ++j) {
        float p = __expf(sv[j] - mnew);
        ssum += p;
        Ps[lr][lseg + j] = (__bf16)(p * crow[j]);
      }
      ssum += __shfl_xor(ssum, 1, 4);
      ssum += __shfl_xor(ssum, 2, 4);
      float scl = __expf(mrow - mnew);
      lrow = lrow * scl + ssum;
      mrow = mnew;
      if ((t & 3) == 0) rowscale[lr] = scl;
    }
    __syncthreads();

    // rescale accumulators, then O += P V
#pragma unroll
    for (int c = 0; c < 2; ++c)
#pragma unroll
      for (int r = 0; r < 8; ++r)
        oacc[c][r] *= rowscale[rt * 16 + half * 8 + r];
#pragma unroll
    for (int kk = 0; kk < 64; kk += 32) {
      bf16x16 af = frag_a(&Ps[0][0], 72, rt * 16 + ln, half, kk);
#pragma unroll
      for (int c = 0; c < 2; ++c) {
        int dt = ct0 + c;
        bf16x16 bf = frag_b(&Vt[0][0], 72, dt * 16 + ln, half, kk);
        oacc[c] = wmma_bf16(af, bf, oacc[c]);
      }
    }
  }

  if ((t & 3) == 0) rowl[lr] = lrow;
  __syncthreads();
#pragma unroll
  for (int c = 0; c < 2; ++c) {
    int dt = ct0 + c;
#pragma unroll
    for (int r = 0; r < 8; ++r) {
      int row = rt * 16 + half * 8 + r;
      float v = oacc[c][r] / rowl[row];
      Octx[(size_t)(b * SS + q0 + row) * DD + hh * DKK + dt * 16 + ln] =
          (__bf16)v;
    }
  }
}

// ---------------------------------------------------------------------------
extern "C" void kernel_launch(void* const* d_in, const int* in_sizes, int n_in,
                              void* d_out, int out_size, void* d_ws,
                              size_t ws_size, hipStream_t stream) {
  const float* x     = (const float*)d_in[0];
  const float* prior = (const float*)d_in[1];
  const int*   amask = (const int*)d_in[2];
  const float* Wq = (const float*)d_in[3];   const float* bq = (const float*)d_in[4];
  const float* Wk = (const float*)d_in[5];   const float* bk = (const float*)d_in[6];
  const float* Wv = (const float*)d_in[7];   const float* bv = (const float*)d_in[8];
  const float* Wo = (const float*)d_in[9];   const float* bo = (const float*)d_in[10];
  const float* gWk = (const float*)d_in[11]; const float* gbk = (const float*)d_in[12];
  const float* gWq = (const float*)d_in[13]; const float* gbq = (const float*)d_in[14];
  const float* glng = (const float*)d_in[15]; const float* glnb = (const float*)d_in[16];
  const float* Wi = (const float*)d_in[17];  const float* bi = (const float*)d_in[18];
  const float* Wout = (const float*)d_in[19]; const float* bout = (const float*)d_in[20];
  const float* lng = (const float*)d_in[21]; const float* lnb = (const float*)d_in[22];

  const int R = BB * SS;  // 8192 tokens
  float* out_layer = (float*)d_out;
  float* out_C     = out_layer + (size_t)R * DD;
  float* out_neigh = out_C + (size_t)BB * SS * SS;
  float* out_attn  = out_neigh + (size_t)BB * SS * SS;

  // ---- workspace carve-up (256B aligned blocks) ----
  char* wp = (char*)d_ws;
  auto alloc = [&](size_t bytes) -> void* {
    void* r = (void*)wp;
    wp += (bytes + 255) & ~(size_t)255;
    return r;
  };
  const size_t DxD = (size_t)DD * DD, DxI = (size_t)DD * DII, RD = (size_t)R * DD;
  __bf16* WtWq  = (__bf16*)alloc(DxD * 2);
  __bf16* WtWk  = (__bf16*)alloc(DxD * 2);
  __bf16* WtWv  = (__bf16*)alloc(DxD * 2);
  __bf16* WtWo  = (__bf16*)alloc(DxD * 2);
  __bf16* WtgWq = (__bf16*)alloc(DxD * 2);
  __bf16* WtgWk = (__bf16*)alloc(DxD * 2);
  __bf16* WtWi  = (__bf16*)alloc(DxI * 2);   // [3072][768]
  __bf16* WtWou = (__bf16*)alloc(DxI * 2);   // [768][3072]
  __bf16* xb    = (__bf16*)alloc(RD * 2);    // bf16 hidden states
  __bf16* hb    = (__bf16*)alloc(RD * 2);    // group LN out -> later v_lin
  __bf16* b1b   = (__bf16*)alloc(RD * 2);    // gq -> later q_lin
  __bf16* b2b   = (__bf16*)alloc(RD * 2);    // gk -> later k_lin
  __bf16* ctxb  = (__bf16*)alloc(RD * 2);    // attention context
  __bf16* attnb = (__bf16*)alloc(RD * 2);    // bf16 mirror of attn_out
  float*  preLN = (float*)alloc(RD * 4);     // FFN output before final LN
  float*  dpl   = (float*)alloc((size_t)R * 4);
  float*  dmi   = (float*)alloc((size_t)R * 4);
  float*  ebuf  = (float*)alloc((size_t)R * 4);
  float*  Pbuf  = (float*)alloc((size_t)R * 4);
  // FFN intermediate (bf16), chunked by remaining workspace
  size_t fixed = (size_t)(wp - (char*)d_ws);
  size_t avail = ws_size > fixed ? ws_size - fixed : 0;
  int chunk = (int)(avail / ((size_t)DII * 2));
  chunk = (chunk / 128) * 128;
  if (chunk > R) chunk = R;
  if (chunk < 128) chunk = 128;
  __bf16* interb = (__bf16*)alloc((size_t)chunk * DII * 2);

  dim3 blk(256);
  // 0) one-time weight transpose/convert + x conversion
  transpose_to_bf16<<<dim3(DD / 32, DD / 32), blk, 0, stream>>>(Wq, WtWq, DD, DD);
  transpose_to_bf16<<<dim3(DD / 32, DD / 32), blk, 0, stream>>>(Wk, WtWk, DD, DD);
  transpose_to_bf16<<<dim3(DD / 32, DD / 32), blk, 0, stream>>>(Wv, WtWv, DD, DD);
  transpose_to_bf16<<<dim3(DD / 32, DD / 32), blk, 0, stream>>>(Wo, WtWo, DD, DD);
  transpose_to_bf16<<<dim3(DD / 32, DD / 32), blk, 0, stream>>>(gWq, WtgWq, DD, DD);
  transpose_to_bf16<<<dim3(DD / 32, DD / 32), blk, 0, stream>>>(gWk, WtgWk, DD, DD);
  transpose_to_bf16<<<dim3(DII / 32, DD / 32), blk, 0, stream>>>(Wi, WtWi, DD, DII);
  transpose_to_bf16<<<dim3(DD / 32, DII / 32), blk, 0, stream>>>(Wout, WtWou, DII, DD);
  f32_to_bf16_vec4<<<(int)(RD / 1024), blk, 0, stream>>>(x, xb);

  // 1) group LayerNorm: h = LN(x) (bf16 only)
  layernorm_rows<<<R / 8, blk, 0, stream>>>(x, glng, glnb, nullptr, hb);
  // 2) group q/k projections (bf16 only)
  gemm_bf16_wmma<<<dim3(DD / 128, R / 128), blk, 0, stream>>>(
      hb, WtgWq, gbq, nullptr, nullptr, b1b, R, DD, DD, 0);
  gemm_bf16_wmma<<<dim3(DD / 128, R / 128), blk, 0, stream>>>(
      hb, WtgWk, gbk, nullptr, nullptr, b2b, R, DD, DD, 0);
  // 3) neighbor dots + per-batch softmax / prefix-log scan
  neighbor_dots<<<R / 8, blk, 0, stream>>>(b1b, b2b, dpl, dmi);
  group_scan<<<BB, 512, 0, stream>>>(dpl, dmi, amask, prior, ebuf, Pbuf);
  // 4) dense C / neigh outputs
  fill_c_neigh<<<(BB * SS * SS) / 256, blk, 0, stream>>>(prior, ebuf, Pbuf,
                                                         out_C, out_neigh);
  // 5) MHA q/k/v projections (bf16 only, buffers reused)
  gemm_bf16_wmma<<<dim3(DD / 128, R / 128), blk, 0, stream>>>(
      xb, WtWq, bq, nullptr, nullptr, b1b, R, DD, DD, 0);
  gemm_bf16_wmma<<<dim3(DD / 128, R / 128), blk, 0, stream>>>(
      xb, WtWk, bk, nullptr, nullptr, b2b, R, DD, DD, 0);
  gemm_bf16_wmma<<<dim3(DD / 128, R / 128), blk, 0, stream>>>(
      xb, WtWv, bv, nullptr, nullptr, hb, R, DD, DD, 0);
  // 6) flash attention with group-prob weighting -> ctx (bf16)
  flash_mha<<<dim3(SS / 64, HH, BB), blk, 0, stream>>>(b1b, b2b, hb, out_C,
                                                       amask, ctxb);
  // 7) output projection -> attn_out (fp32 output #4 + bf16 mirror)
  gemm_bf16_wmma<<<dim3(DD / 128, R / 128), blk, 0, stream>>>(
      ctxb, WtWo, bo, nullptr, out_attn, attnb, R, DD, DD, 0);
  // 8) FFN (chunked): GELU(attn@Wi+bi) @ Wout + bout + attn -> preLN
  for (int r0 = 0; r0 < R; r0 += chunk) {
    int mrows = (R - r0) < chunk ? (R - r0) : chunk;
    gemm_bf16_wmma<<<dim3(DII / 128, mrows / 128), blk, 0, stream>>>(
        attnb + (size_t)r0 * DD, WtWi, bi, nullptr, nullptr, interb,
        mrows, DII, DD, 1);
    gemm_bf16_wmma<<<dim3(DD / 128, mrows / 128), blk, 0, stream>>>(
        interb, WtWou, bout, out_attn + (size_t)r0 * DD,
        preLN + (size_t)r0 * DD, nullptr, mrows, DD, DII, 0);
  }
  // 9) final LayerNorm -> layer_out (output #1)
  layernorm_rows<<<R / 8, blk, 0, stream>>>(preLN, lng, lnb, out_layer, nullptr);
}
